// InterBandGraphInteraction_16612933501298
// MI455X (gfx1250) — matrix operand order repaired and checked by reading, not compiled, not run on hardware
//
#include <hip/hip_runtime.h>
#include <hip/hip_bf16.h>
#include <math.h>

// ---------------------------------------------------------------------------
// B=16384, N=9, F_IN=1024, BN_F=64, OUT_F=32;  M_total = B*N = 147456
//  Kernel 0: wt[1024][96] bf16 = transpose([Wb;Wg]) (one-shot, L2-resident)
//  Kernel 1: Y[147456,96] = X @ [Wb;Wg]^T via v_wmma_f32_16x16x32_bf16
//            cols 0..63 -> xa = tanh(.+bb), cols 64..95 -> h
//            W chunk staged to LDS via GLOBAL_LOAD_ASYNC_TO_LDS_B128
//  Kernel 2: per-batch 9x9 softmax/top2/self-loop/normalize + adj@h+bg, relu
// ---------------------------------------------------------------------------

typedef __attribute__((ext_vector_type(16))) __bf16 v16bf;
typedef __attribute__((ext_vector_type(8)))  __bf16 v8bf;
typedef __attribute__((ext_vector_type(4)))  __bf16 v4bf;
typedef __attribute__((ext_vector_type(8)))  float  v8f;
typedef int v4i __attribute__((vector_size(4 * sizeof(int))));
typedef __attribute__((address_space(1))) v4i* gas_v4i_ptr;
typedef __attribute__((address_space(3))) v4i* las_v4i_ptr;

#define MTOT   147456
#define KDIM   1024
#define MBLK   128         // rows per block (8 waves x 16)
#define KC     64          // K-chunk staged in LDS
#define XSTR   72          // bf16/LDS X row (pad; 144B, 16B aligned)
#define WSTR   104         // bf16/LDS W row (pad; 208B, 16B aligned)

__device__ __forceinline__ v16bf cat8(v8bf lo, v8bf hi) {
    return __builtin_shufflevector(lo, hi, 0,1,2,3,4,5,6,7,8,9,10,11,12,13,14,15);
}

__device__ __forceinline__ float fast_tanhf(float x) {
    // tanh(x) = 1 - 2/(e^(2x)+1); saturates correctly at +/-inf via v_exp_f32
    const float t = __expf(2.0f * x);
    return 1.0f - 2.0f / (t + 1.0f);
}

// ---- async global->LDS 16B copy (ASYNCcnt path), with fallbacks ----
__device__ __forceinline__ void async_copy16(const void* g, void* lds) {
#if __has_builtin(__builtin_amdgcn_global_load_async_to_lds_b128)
    __builtin_amdgcn_global_load_async_to_lds_b128(
        (gas_v4i_ptr)(size_t)g,
        (las_v4i_ptr)(unsigned)(size_t)lds,   // low 32 bits = LDS byte addr
        0, 0);
#else
    const unsigned loff = (unsigned)(size_t)lds;
    asm volatile("global_load_async_to_lds_b128 %0, %1, off"
                 :: "v"(loff), "v"(g) : "memory");
#endif
}
__device__ __forceinline__ void wait_async0() {
#if __has_builtin(__builtin_amdgcn_s_wait_asynccnt)
    __builtin_amdgcn_s_wait_asynccnt(0);
#else
    asm volatile("s_wait_asynccnt 0x0" ::: "memory");
#endif
}

// ---------------------------------------------------------------------------
// Kernel 0: wt[k*96 + n] = (bf16) W[n][k],  W = [Wb(64 rows); Wg(32 rows)]
// ---------------------------------------------------------------------------
__global__ __launch_bounds__(256)
void ibgi_wprep(const float* __restrict__ Wb, const float* __restrict__ Wg,
                __bf16* __restrict__ wt)
{
    const int idx = blockIdx.x * 256 + threadIdx.x;    // 0..49151 (2 elems each)
    const int k = idx / 48;
    const int j = idx % 48;
    const int n0 = j * 2, n1 = j * 2 + 1;
    const float a = (n0 < 64) ? Wb[(size_t)n0 * KDIM + k] : Wg[(size_t)(n0 - 64) * KDIM + k];
    const float b = (n1 < 64) ? Wb[(size_t)n1 * KDIM + k] : Wg[(size_t)(n1 - 64) * KDIM + k];
    __bf16* p = wt + (size_t)k * 96 + n0;
    p[0] = (__bf16)a;
    p[1] = (__bf16)b;
}

// ---------------------------------------------------------------------------
// Kernel 1: fused GEMM + activation
// ---------------------------------------------------------------------------
__global__ __launch_bounds__(256)
void ibgi_gemm_act(const float* __restrict__ x,
                   const __bf16* __restrict__ wt,
                   const float* __restrict__ bb,
                   float* __restrict__ xa,
                   float* __restrict__ h)
{
    __shared__ __align__(16) __bf16 xlds[MBLK * XSTR];   // [row][k]
    __shared__ __align__(16) __bf16 wlds[KC * WSTR];     // [k][n] (pre-transposed)

    const int t     = threadIdx.x;
    const int lane  = t & 31;
    const int wv    = t >> 5;            // wave id -> M-tile
    const int half  = lane >> 4;
    const int mBase = blockIdx.x * MBLK;

    const v8f zero8 = {0.f,0.f,0.f,0.f,0.f,0.f,0.f,0.f};
    v8f acc[6];
    #pragma unroll
    for (int i = 0; i < 6; ++i) acc[i] = zero8;

    for (int kc = 0; kc < KDIM / KC; ++kc) {
        __syncthreads();   // previous chunk's readers have drained (dscnt==0)

        // ---- W chunk: async copy wt rows [kc*64, kc*64+64) -> wlds ----
        #pragma unroll
        for (int i = 0; i < 3; ++i) {
            const int idx = i * 256 + t;          // 0..767 16B chunks
            const int row = idx / 12;
            const int cj  = idx % 12;
            async_copy16(wt + (size_t)(kc * KC + row) * 96 + cj * 8,
                         wlds + row * WSTR + cj * 8);
        }
        // ---- X chunk: 128 rows x 64 fp32 -> bf16 LDS (convert in VGPRs) ----
        #pragma unroll
        for (int i = 0; i < 8; ++i) {
            const int idx = i * 256 + t;          // float4 index 0..2047
            const int r   = idx >> 4;
            const int kq  = idx & 15;
            const float4 v = *(const float4*)(x + (size_t)(mBase + r) * KDIM + kc * KC + kq * 4);
            v4bf o;
            o[0] = (__bf16)v.x; o[1] = (__bf16)v.y; o[2] = (__bf16)v.z; o[3] = (__bf16)v.w;
            *(v4bf*)(xlds + r * XSTR + kq * 4) = o;
        }
        if (kc + 1 < KDIM / KC) {   // speculative prefetch of next X chunk
            __builtin_prefetch(x + (size_t)(mBase + (t >> 1)) * KDIM + (kc + 1) * KC + (t & 1) * 32, 0, 0);
        }
        wait_async0();
        __syncthreads();

        // ---- compute: 6 N-tiles x 2 K-steps = 12 WMMA per wave per chunk ----
        const __bf16* arow = xlds + (wv * 16 + (lane & 15)) * XSTR;
        #pragma unroll
        for (int ks = 0; ks < 2; ++ks) {
            const int kk = ks * 32;
            const v8bf alo = *(const v8bf*)(arow + kk + half * 8);
            const v8bf ahi = *(const v8bf*)(arow + kk + half * 8 + 16);
            const v16bf afrag = cat8(alo, ahi);
            const __bf16* brow = wlds + kk * WSTR + (lane & 15);   // lane%16 = N col
            #pragma unroll
            for (int nt = 0; nt < 6; ++nt) {
                v16bf bfrag;
                #pragma unroll
                for (int s = 0; s < 8; ++s) {
                    bfrag[s]     = brow[(half * 8 + s) * WSTR + nt * 16];
                    bfrag[s + 8] = brow[(16 + half * 8 + s) * WSTR + nt * 16];
                }
                acc[nt] = __builtin_amdgcn_wmma_f32_16x16x32_bf16(
                    false, afrag, false, bfrag, (short)0, acc[nt], false, false);
            }
        }
    }

    // ---- epilogue: D layout lane(0-15)=N col, VGPR r -> M=r (+8 lanes 16-31) ----
    const int col0 = lane & 15;
    #pragma unroll
    for (int nt = 0; nt < 6; ++nt) {
        const int col = nt * 16 + col0;
        #pragma unroll
        for (int r = 0; r < 8; ++r) {
            const int row = mBase + wv * 16 + half * 8 + r;
            const float v = acc[nt][r];
            if (nt < 4) {
                xa[(size_t)row * 64 + col] = fast_tanhf(v + bb[col]);
            } else {
                h[(size_t)row * 32 + (col - 64)] = v;
            }
        }
    }
}

// ---------------------------------------------------------------------------
// Kernel 2: one wave per batch (8 batches per 256-thread block)
// ---------------------------------------------------------------------------
__global__ __launch_bounds__(256)
void ibgi_graph(const float* __restrict__ xa,
                const float* __restrict__ h,
                const float* __restrict__ bg,
                float* __restrict__ out)
{
    __shared__ __align__(16) float lxa [8][9 * 64];
    __shared__ float lsc [8][9 * 12];
    __shared__ float ladj[8][9 * 12];
    __shared__ float ld  [8][12];

    const int t    = threadIdx.x;
    const int lane = t & 31;
    const int wv   = t >> 5;
    const int batch = blockIdx.x * 8 + wv;

    const float* xab = xa + (size_t)batch * 9 * 64;
    const float* hb  = h  + (size_t)batch * 9 * 32;

    for (int i = 0; i < 5; ++i) {
        const int idx = i * 32 + lane;
        if (idx < 144) ((float4*)lxa[wv])[idx] = ((const float4*)xab)[idx];
    }
    float hreg[9];
    #pragma unroll
    for (int m = 0; m < 9; ++m) hreg[m] = hb[m * 32 + lane];   // lane = out channel
    const float bgv = bg[lane];
    __syncthreads();

    for (int p = lane; p < 81; p += 32) {
        const int n = p / 9, m = p % 9;
        const float* an = lxa[wv] + n * 64;
        const float* am = lxa[wv] + m * 64;
        float s = 0.f;
        #pragma unroll 8
        for (int f = 0; f < 64; ++f) s = fmaf(an[f], am[f], s);
        lsc[wv][n * 12 + m] = s;
    }
    __syncthreads();

    if (lane < 9) {
        const int n = lane;
        float s[9];
        #pragma unroll
        for (int i = 0; i < 9; ++i) s[i] = lsc[wv][n * 12 + i];
        float mx = s[0];
        #pragma unroll
        for (int i = 1; i < 9; ++i) mx = fmaxf(mx, s[i]);
        float e[9], sum = 0.f;
        #pragma unroll
        for (int i = 0; i < 9; ++i) { e[i] = __expf(s[i] - mx); sum += e[i]; }
        const float inv = 1.f / sum;
        float p_[9];
        #pragma unroll
        for (int i = 0; i < 9; ++i) p_[i] = e[i] * inv;
        int i1 = 0; float v1 = p_[0];
        #pragma unroll
        for (int i = 1; i < 9; ++i) if (p_[i] > v1) { v1 = p_[i]; i1 = i; }
        int i2 = -1; float v2 = -1e30f;
        #pragma unroll
        for (int i = 0; i < 9; ++i) if (i != i1 && p_[i] > v2) { v2 = p_[i]; i2 = i; }
        float r[9]; float sumr = 0.f;
        #pragma unroll
        for (int i = 0; i < 9; ++i) r[i] = (i == i1 || i == i2) ? p_[i] : 0.f;
        r[n] = 1.f;                                 // self loop (after mask)
        #pragma unroll
        for (int i = 0; i < 9; ++i) sumr += r[i];
        ld[wv][n] = rsqrtf(fmaxf(sumr, 1.f));       // clip(deg,1)^-0.5
        #pragma unroll
        for (int i = 0; i < 9; ++i) ladj[wv][n * 12 + i] = r[i];
    }
    __syncthreads();

    #pragma unroll
    for (int n = 0; n < 9; ++n) {
        const float dn = ld[wv][n];
        float accv = 0.f;
        #pragma unroll
        for (int m = 0; m < 9; ++m) {
            const float a = dn * ladj[wv][n * 12 + m] * ld[wv][m];
            accv = fmaf(a, hreg[m], accv);
        }
        out[(size_t)batch * 288 + n * 32 + lane] = fmaxf(accv + bgv, 0.f);
    }
}

extern "C" void kernel_launch(void* const* d_in, const int* in_sizes, int n_in,
                              void* d_out, int out_size, void* d_ws, size_t ws_size,
                              hipStream_t stream) {
    const float* x  = (const float*)d_in[0];
    const float* Wb = (const float*)d_in[1];
    const float* bb = (const float*)d_in[2];
    const float* Wg = (const float*)d_in[3];
    const float* bg = (const float*)d_in[4];

    float*  xa  = (float*)d_ws;                                  // 147456 x 64 fp32
    float*  h   = xa + (size_t)MTOT * 64;                        // 147456 x 32 fp32
    __bf16* wt  = (__bf16*)((char*)d_ws + (size_t)MTOT * 96 * 4);// 1024 x 96 bf16
    float*  out = (float*)d_out;                                 // 16384 x 9 x 32 fp32

    ibgi_wprep   <<<192,          256, 0, stream>>>(Wb, Wg, wt);
    ibgi_gemm_act<<<MTOT / MBLK,  256, 0, stream>>>(x, wt, bb, xa, h);
    ibgi_graph   <<<16384 / 8,    256, 0, stream>>>(xa, h, bg, out);
}
